// ACAM_3934190043363
// MI455X (gfx1250) — compile-verified, matrix operand order checked by reading, not run on hardware
//
#include <hip/hip_runtime.h>
#include <hip/hip_bf16.h>
#include <math.h>

// ---------------------------------------------------------------------------
// ACAM fused forward for MI455X (gfx1250, wave32, WMMA bf16 16x16x32).
// One wave = 16 batch rows for the whole 7-step recurrence; weights pre-packed
// into WMMA B-fragment order (BN scale folded) by tiny prep kernels into d_ws.
// v2: register-pressure oriented restructure (N-outer GEMM loops, unroll
// control, biases staged in LDS) to kill scratch spills seen in v1 asm.
// ---------------------------------------------------------------------------

typedef __bf16 bf16;
typedef __bf16 bf16x16 __attribute__((ext_vector_type(16)));
typedef float  floatx8 __attribute__((ext_vector_type(8)));

#define ACAM_EPS 1e-5f

// Packed-weight offsets in d_ws (bf16 elements). Fragment = 512 bf16 elems,
// layout: frag = nTile*KC + kc ; element addr = frag*512 + lane*16 + slot.
enum : int {
  PK_ATT1 = 0,                      //  7x128  KC=1  NT=8
  PK_ATT2 = PK_ATT1 + 1 * 8 * 512,  // 128x128 KC=4  NT=8
  PK_INP1 = PK_ATT2 + 4 * 8 * 512,  // 560x128 KC=18 NT=8
  PK_INP2 = PK_INP1 + 18 * 8 * 512, // 128x128 KC=4  NT=8
  PK_WIH  = PK_INP2 + 4 * 8 * 512,  // 128x512 KC=4  NT=32
  PK_WHH  = PK_WIH + 4 * 32 * 512,  // 128x512 KC=4  NT=32
  PK_DEC  = PK_WHH + 4 * 32 * 512,  // 128x7   KC=4  NT=1
  PK_ACT1 = PK_DEC + 4 * 1 * 512,   // 128x256 KC=4  NT=16
  PK_ACT2 = PK_ACT1 + 4 * 16 * 512, // 256x256 KC=8  NT=16
  PK_CLS  = PK_ACT2 + 8 * 16 * 512, // 256x7   KC=8  NT=1
  PK_TOTAL = PK_CLS + 8 * 1 * 512,  // 346112 bf16 elems
};
// Folded-bias offsets (floats), placed after packed weights in d_ws.
enum : int {
  FB_ATT1 = 0, FB_ATT2 = 128, FB_INP1 = 256, FB_INP2 = 384,
  FB_LSTM = 512,               // 512 floats: (b_ih + b_hh), gate order i,f,g,o
  FB_DEC  = 1024,              // 16 (padded)
  FB_ACT1 = 1040, FB_ACT2 = 1296,
  FB_CLS  = 1552,              // 16 (padded)
  FB_TOTAL = 1568,
};

__device__ __forceinline__ float sigf(float x) { return 1.f / (1.f + __expf(-x)); }

__device__ __forceinline__ floatx8 wmma_bf16(bf16x16 a, bf16x16 b, floatx8 c) {
  return __builtin_amdgcn_wmma_f32_16x16x32_bf16(false, a, false, b, (short)0, c,
                                                 false, false);
}

// A-fragment from a row-major bf16 LDS tile (ld = row stride in elems).
// ISA 16-bit A layout: lanes 0-15 -> M=lane, K slots {0..7,16..23};
// lanes 16-31 -> M=lane-16, K slots {8..15,24..31}. Two b128 LDS loads.
__device__ __forceinline__ bf16x16 frag_lds(const bf16* buf, int ld, int kc) {
  int lane = threadIdx.x & 31;
  const bf16* p = buf + (lane & 15) * ld + kc * 32 + ((lane >> 4) << 3);
  bf16x16 r;
  ((uint4*)&r)[0] = *(const uint4*)p;         // K slots 0..7  (+lane-half offset)
  ((uint4*)&r)[1] = *(const uint4*)(p + 16);  // K slots 16..23 (+lane-half offset)
  return r;
}

// B-fragment from pre-packed global memory: contiguous 32B per lane.
__device__ __forceinline__ bf16x16 frag_glb(const bf16* pk, int frag) {
  int lane = threadIdx.x & 31;
  const bf16* p = pk + (size_t)frag * 512 + lane * 16;
  bf16x16 r;
  ((uint4*)&r)[0] = *(const uint4*)p;
  ((uint4*)&r)[1] = *(const uint4*)(p + 8);
  return r;
}

// ---------------------------------------------------------------------------
// Prep kernels: pack W (N,K fp32 row-major) into B-fragment bf16 order with
// BN scale folded; fold biases.
// ---------------------------------------------------------------------------
__global__ void pack_w(const float* __restrict__ Wm, const float* __restrict__ g,
                       const float* __restrict__ var, bf16* __restrict__ dst,
                       int K, int N, int KC, int NT) {
  int idx = blockIdx.x * blockDim.x + threadIdx.x;
  int total = KC * NT * 512;
  if (idx >= total) return;
  int slot = idx & 15;
  int lane = (idx >> 4) & 31;
  int frag = idx >> 9;
  int kc = frag % KC;
  int nTile = frag / KC;
  int kl = (slot & 7) + ((slot >> 3) << 4) + ((lane >> 4) << 3);
  int k = kc * 32 + kl;
  int n = nTile * 16 + (lane & 15);
  float v = 0.f;
  if (k < K && n < N) {
    float s = 1.f;
    if (g) s = g[n] * rsqrtf(var[n] + ACAM_EPS);
    v = Wm[n * K + k] * s;
  }
  dst[idx] = (bf16)v;
}

__global__ void fold_b(const float* __restrict__ b, const float* __restrict__ g,
                       const float* __restrict__ beta, const float* __restrict__ mean,
                       const float* __restrict__ var, float* __restrict__ dst,
                       int N, int Npad) {
  int n = blockIdx.x * blockDim.x + threadIdx.x;
  if (n >= Npad) return;
  float v = 0.f;
  if (n < N) {
    if (g) {
      float s = g[n] * rsqrtf(var[n] + ACAM_EPS);
      v = b[n] * s + (beta[n] - mean[n] * s);
    } else {
      v = b[n];
    }
  }
  dst[n] = v;
}

__global__ void sum_b(const float* __restrict__ a, const float* __restrict__ b,
                      float* __restrict__ dst, int N) {
  int n = blockIdx.x * blockDim.x + threadIdx.x;
  if (n < N) dst[n] = a[n] + b[n];
}

// ---------------------------------------------------------------------------
// Main fused kernel: one wave (32 threads) per block, 16 batch rows per wave.
// ---------------------------------------------------------------------------
struct WaveLds {
  alignas(16) float attn[16 * 8];      // attention (7, padded to 8)
  alignas(16) float rowStats[16 * 4];  // h_mu, h_rs, c_mu, c_rs per row
  alignas(16) float fbL[FB_TOTAL];     // folded biases (LDS-resident)
  alignas(16) float lnG[128];          // LayerNorm gamma
  alignas(16) float lnB[128];          // LayerNorm beta
  alignas(16) float hS[2048];          // h state fp32 (16x128)
  alignas(16) float cS[2048];          // c state fp32
  alignas(16) bf16 hB[2048];           // bf16(LN(h)) staging for gates GEMM
  alignas(16) bf16 aggB[2048];         // agg bf16
  alignas(16) bf16 midB[2048];         // enc_inp mid / bf16(h_new) staging
  alignas(16) bf16 amidB[2048];        // enc_att mid
  alignas(16) bf16 attB[512];          // 16x32 zero-padded attention tile
  alignas(16) bf16 attd[16 * 576];     // attended (K padded 560->576); reused by head
};

__global__ __launch_bounds__(32) void acam_main(
    const float* __restrict__ feat, const bf16* __restrict__ pk,
    const float* __restrict__ fb, const float* __restrict__ ln_g,
    const float* __restrict__ ln_b, float* __restrict__ out) {
  __shared__ WaveLds w;
  const int lane = threadIdx.x & 31;
  const int nl = lane & 15;
  const int mB = (lane >> 4) << 3;
  const int rowBase = blockIdx.x * 16;

  // one-time staging: zero state, biases + LN params into LDS
  for (int i = lane; i < 128; i += 32) w.attn[i] = 0.f;
  for (int i = lane; i < 2048; i += 32) { w.hS[i] = 0.f; w.cS[i] = 0.f; }
  for (int i = lane; i < FB_TOTAL; i += 32) w.fbL[i] = fb[i];
  for (int i = lane; i < 128; i += 32) { w.lnG[i] = ln_g[i]; w.lnB[i] = ln_b[i]; }

#pragma unroll 1
  for (int step = 0; step < 7; ++step) {
    // ---- stage attention into zero-padded 16x32 bf16 tile
    for (int i = lane; i < 512; i += 32) {
      int m = i >> 5, k = i & 31;
      w.attB[i] = (bf16)((k < 7) ? w.attn[m * 8 + k] : 0.f);
    }
    // ---- branch1 layer1: amid = relu(att @ W1' + b1')
    {
      bf16x16 a = frag_lds(w.attB, 32, 0);
#pragma unroll 1
      for (int nt = 0; nt < 8; ++nt) {
        floatx8 acc = {};
        acc = wmma_bf16(a, frag_glb(pk + PK_ATT1, nt), acc);
        int n = nt * 16 + nl;
        float bias = w.fbL[FB_ATT1 + n];
#pragma unroll
        for (int r = 0; r < 8; ++r)
          w.amidB[(mB + r) * 128 + n] = (bf16)fmaxf(acc[r] + bias, 0.f);
      }
    }
    // ---- attended = features * attention, staged 16x576 bf16 (coalesced)
#pragma unroll 1
    for (int m = 0; m < 16; ++m) {
      const float* fr = feat + (size_t)(rowBase + m) * 560;
#pragma unroll 1
      for (int k = lane; k < 576; k += 32) {
        float v = 0.f;
        if (k < 560) v = fr[k] * w.attn[m * 8 + k / 80];
        w.attd[m * 576 + k] = (bf16)v;
      }
    }
    // ---- branch2 layer1: mid = relu(attended @ Winp1' + b); N-outer, K-inner
#pragma unroll 1
    for (int nt = 0; nt < 8; ++nt) {
      floatx8 acc = {};
#pragma unroll 2
      for (int kc = 0; kc < 18; ++kc)
        acc = wmma_bf16(frag_lds(w.attd, 576, kc),
                        frag_glb(pk + PK_INP1, nt * 18 + kc), acc);
      int n = nt * 16 + nl;
      float bias = w.fbL[FB_INP1 + n];
#pragma unroll
      for (int r = 0; r < 8; ++r)
        w.midB[(mB + r) * 128 + n] = (bf16)fmaxf(acc[r] + bias, 0.f);
    }
    // ---- layer2 both branches fused: agg = relu((amid@W2a'+b) + (mid@W2x'+b))
#pragma unroll 1
    for (int nt = 0; nt < 8; ++nt) {
      floatx8 aA = {}, aX = {};
#pragma unroll
      for (int kc = 0; kc < 4; ++kc) {
        aA = wmma_bf16(frag_lds(w.amidB, 128, kc), frag_glb(pk + PK_ATT2, nt * 4 + kc), aA);
        aX = wmma_bf16(frag_lds(w.midB, 128, kc), frag_glb(pk + PK_INP2, nt * 4 + kc), aX);
      }
      int n = nt * 16 + nl;
      float ba = w.fbL[FB_ATT2 + n], bx = w.fbL[FB_INP2 + n];
#pragma unroll
      for (int r = 0; r < 8; ++r) {
        float v = (aA[r] + ba) + (aX[r] + bx);
        w.aggB[(mB + r) * 128 + n] = (bf16)fmaxf(v, 0.f);
      }
    }
    // ---- LayerNorm stats for h,c (step>0), then bf16 staging of h_in
    if (step > 0) {
      int row = nl, half = lane >> 4;
      float sh = 0, sh2 = 0, sc = 0, sc2 = 0;
#pragma unroll 1
      for (int j = 0; j < 64; ++j) {
        float hv = w.hS[row * 128 + half * 64 + j]; sh += hv; sh2 += hv * hv;
        float cv = w.cS[row * 128 + half * 64 + j]; sc += cv; sc2 += cv * cv;
      }
      sh += __shfl_xor(sh, 16); sh2 += __shfl_xor(sh2, 16);
      sc += __shfl_xor(sc, 16); sc2 += __shfl_xor(sc2, 16);
      if (half == 0) {
        float mu = sh * (1.f / 128.f), va = sh2 * (1.f / 128.f) - mu * mu;
        w.rowStats[row * 4 + 0] = mu;
        w.rowStats[row * 4 + 1] = rsqrtf(va + ACAM_EPS);
        mu = sc * (1.f / 128.f); va = sc2 * (1.f / 128.f) - mu * mu;
        w.rowStats[row * 4 + 2] = mu;
        w.rowStats[row * 4 + 3] = rsqrtf(va + ACAM_EPS);
      }
    }
#pragma unroll 1
    for (int i = lane; i < 2048; i += 32) {
      int m = i >> 7, n = i & 127;
      float hv = w.hS[i];
      if (step > 0)
        hv = (hv - w.rowStats[m * 4]) * w.rowStats[m * 4 + 1] * w.lnG[n] + w.lnB[n];
      w.hB[i] = (bf16)hv;
    }
    // ---- LSTM gates, one hidden 16-tile at a time; 4 gate accumulators live
#pragma unroll 1
    for (int t = 0; t < 8; ++t) {
      floatx8 gi = {}, gf = {}, gg = {}, go = {};
#pragma unroll 1
      for (int kc = 0; kc < 4; ++kc) {
        bf16x16 a = frag_lds(w.aggB, 128, kc);
        gi = wmma_bf16(a, frag_glb(pk + PK_WIH, (t) * 4 + kc), gi);
        gf = wmma_bf16(a, frag_glb(pk + PK_WIH, (t + 8) * 4 + kc), gf);
        gg = wmma_bf16(a, frag_glb(pk + PK_WIH, (t + 16) * 4 + kc), gg);
        go = wmma_bf16(a, frag_glb(pk + PK_WIH, (t + 24) * 4 + kc), go);
      }
#pragma unroll 1
      for (int kc = 0; kc < 4; ++kc) {
        bf16x16 a = frag_lds(w.hB, 128, kc);
        gi = wmma_bf16(a, frag_glb(pk + PK_WHH, (t) * 4 + kc), gi);
        gf = wmma_bf16(a, frag_glb(pk + PK_WHH, (t + 8) * 4 + kc), gf);
        gg = wmma_bf16(a, frag_glb(pk + PK_WHH, (t + 16) * 4 + kc), gg);
        go = wmma_bf16(a, frag_glb(pk + PK_WHH, (t + 24) * 4 + kc), go);
      }
      int n = t * 16 + nl;
      float bi = w.fbL[FB_LSTM + n], bf2 = w.fbL[FB_LSTM + 128 + n];
      float bg = w.fbL[FB_LSTM + 256 + n], bo = w.fbL[FB_LSTM + 384 + n];
#pragma unroll
      for (int r = 0; r < 8; ++r) {
        int m = mB + r;
        float cin = w.cS[m * 128 + n];
        if (step > 0)
          cin = (cin - w.rowStats[m * 4 + 2]) * w.rowStats[m * 4 + 3] * w.lnG[n] + w.lnB[n];
        float ig = sigf(gi[r] + bi);
        float fg = sigf(gf[r] + bf2);
        float gv = tanhf(gg[r] + bg);
        float og = sigf(go[r] + bo);
        float cn = fg * cin + ig * gv;
        float hn = og * tanhf(cn);
        w.cS[m * 128 + n] = cn;
        w.hS[m * 128 + n] = hn;
        w.midB[m * 128 + n] = (bf16)hn;  // bf16(h_new) for dec / next head
      }
    }
    // ---- decoder + smooth softmax (not on last step)
    if (step < 6) {
      floatx8 acc = {};
#pragma unroll
      for (int kc = 0; kc < 4; ++kc)
        acc = wmma_bf16(frag_lds(w.midB, 128, kc), frag_glb(pk + PK_DEC, kc), acc);
      if (nl < 8) {
        float bias = w.fbL[FB_DEC + nl];
#pragma unroll
        for (int r = 0; r < 8; ++r) {
          float s = (nl < 7) ? sigf(acc[r] + bias) : 0.f;
          w.attn[(mB + r) * 8 + nl] = s;
        }
      }
      if (lane < 16) {
        float s = 0.f;
#pragma unroll
        for (int j = 0; j < 7; ++j) s += w.attn[lane * 8 + j];
        float inv = 1.f / s;
#pragma unroll
        for (int j = 0; j < 7; ++j) w.attn[lane * 8 + j] *= inv;
      }
    }
  }

  // ---- action head: relu(bn(act1(h))), relu(bn(act2(.))), sigmoid(cls(.))
  bf16* actB1 = w.attd;            // 16x256 bf16
  bf16* actB2 = w.attd + 16 * 256; // 16x256 bf16 (attd is 16x576: both fit)
#pragma unroll 1
  for (int nt = 0; nt < 16; ++nt) {
    floatx8 acc = {};
#pragma unroll
    for (int kc = 0; kc < 4; ++kc)
      acc = wmma_bf16(frag_lds(w.midB, 128, kc), frag_glb(pk + PK_ACT1, nt * 4 + kc), acc);
    int n = nt * 16 + nl;
    float bias = w.fbL[FB_ACT1 + n];
#pragma unroll
    for (int r = 0; r < 8; ++r)
      actB1[(mB + r) * 256 + n] = (bf16)fmaxf(acc[r] + bias, 0.f);
  }
#pragma unroll 1
  for (int nt = 0; nt < 16; ++nt) {
    floatx8 acc = {};
#pragma unroll 2
    for (int kc = 0; kc < 8; ++kc)
      acc = wmma_bf16(frag_lds(actB1, 256, kc), frag_glb(pk + PK_ACT2, nt * 8 + kc), acc);
    int n = nt * 16 + nl;
    float bias = w.fbL[FB_ACT2 + n];
#pragma unroll
    for (int r = 0; r < 8; ++r)
      actB2[(mB + r) * 256 + n] = (bf16)fmaxf(acc[r] + bias, 0.f);
  }
  {
    floatx8 acc = {};
#pragma unroll 2
    for (int kc = 0; kc < 8; ++kc)
      acc = wmma_bf16(frag_lds(actB2, 256, kc), frag_glb(pk + PK_CLS, kc), acc);
    if (nl < 7) {
      float bias = w.fbL[FB_CLS + nl];
#pragma unroll
      for (int r = 0; r < 8; ++r)
        out[(size_t)(rowBase + mB + r) * 7 + nl] = sigf(acc[r] + bias);
    }
  }
}

// ---------------------------------------------------------------------------
extern "C" void kernel_launch(void* const* d_in, const int* in_sizes, int n_in,
                              void* d_out, int out_size, void* d_ws, size_t ws_size,
                              hipStream_t stream) {
  (void)in_sizes; (void)n_in; (void)out_size; (void)ws_size;
  auto in = [&](int i) { return (const float*)d_in[i]; };
  // d_in order (setup_inputs dict order):
  // 0 features | 1 att1.w 2 att1.b | 3..6 att_bn1 g,b,m,v | 7 att2.w 8 att2.b
  // 9..12 att_bn2 | 13 inp1.w 14 inp1.b | 15..18 inp_bn1 | 19 inp2.w 20 inp2.b
  // 21..24 inp_bn2 | 25 dec.w 26 dec.b | 27..30 dec_bn | 31 w_ih 32 w_hh
  // 33 b_ih 34 b_hh | 35 ln.g 36 ln.b | 37 act1.w 38 act1.b | 39..42 act1_bn
  // 43 act2.w 44 act2.b | 45..48 act2_bn | 49 cls.w 50 cls.b
  bf16* pk = (bf16*)d_ws;
  float* fb = (float*)((char*)d_ws + (size_t)PK_TOTAL * sizeof(bf16));

  auto pack = [&](int off, const float* Wm, const float* g, const float* var,
                  int K, int N, int KC, int NT) {
    int total = KC * NT * 512;
    pack_w<<<(total + 255) / 256, 256, 0, stream>>>(Wm, g, var, pk + off, K, N, KC, NT);
  };
  pack(PK_ATT1, in(1), in(3), in(6), 7, 128, 1, 8);
  pack(PK_ATT2, in(7), in(9), in(12), 128, 128, 4, 8);
  pack(PK_INP1, in(13), in(15), in(18), 560, 128, 18, 8);
  pack(PK_INP2, in(19), in(21), in(24), 128, 128, 4, 8);
  pack(PK_WIH, in(31), nullptr, nullptr, 128, 512, 4, 32);
  pack(PK_WHH, in(32), nullptr, nullptr, 128, 512, 4, 32);
  pack(PK_DEC, in(25), in(27), in(30), 128, 7, 4, 1);
  pack(PK_ACT1, in(37), in(39), in(42), 128, 256, 4, 16);
  pack(PK_ACT2, in(43), in(45), in(48), 256, 256, 8, 16);
  pack(PK_CLS, in(49), nullptr, nullptr, 256, 7, 8, 1);

  auto bias = [&](int off, const float* b, const float* g, const float* be,
                  const float* mean, const float* var, int N, int Npad) {
    fold_b<<<(Npad + 255) / 256, 256, 0, stream>>>(b, g, be, mean, var, fb + off, N, Npad);
  };
  bias(FB_ATT1, in(2), in(3), in(4), in(5), in(6), 128, 128);
  bias(FB_ATT2, in(8), in(9), in(10), in(11), in(12), 128, 128);
  bias(FB_INP1, in(14), in(15), in(16), in(17), in(18), 128, 128);
  bias(FB_INP2, in(20), in(21), in(22), in(23), in(24), 128, 128);
  sum_b<<<2, 256, 0, stream>>>(in(33), in(34), fb + FB_LSTM, 512);
  bias(FB_DEC, in(26), in(27), in(28), in(29), in(30), 7, 16);
  bias(FB_ACT1, in(38), in(39), in(40), in(41), in(42), 256, 256);
  bias(FB_ACT2, in(44), in(45), in(46), in(47), in(48), 256, 256);
  bias(FB_CLS, in(50), nullptr, nullptr, nullptr, nullptr, 7, 16);

  acam_main<<<65536 / 16, 32, 0, stream>>>(in(0), pk, fb, in(35), in(36), (float*)d_out);
}
